// STHD_SpGAT_75814762709195
// MI455X (gfx1250) — compile-verified, matrix operand order checked by reading, not run on hardware
//
#include <hip/hip_runtime.h>
#include <hip/hip_bf16.h>
#include <math.h>

typedef __attribute__((ext_vector_type(16))) _Float16 v16h;
typedef __attribute__((ext_vector_type(8)))  float    v8f;

#define N_CELLS   100000
#define N_CLASSES 50
#define N_GENES   500
#define N_EDGES   600000
#define HID       8
#define NEG_SLOPE 0.2f

#define KT_TILES 16      // 16 * 32 = 512 >= 500 (K padded with zeros)
#define CT_TILES 4       // 4 * 16 = 64  >= 50  (C padded with zeros)
#define ROW_TILES (N_CELLS / 16)   // 6250 exact

// ---------- order-preserving float<->uint encode for atomic segment-max ----------
__device__ __forceinline__ unsigned enc_f(float f) {
    unsigned u = __float_as_uint(f);
    return (u & 0x80000000u) ? ~u : (u | 0x80000000u);
}
__device__ __forceinline__ float dec_f(unsigned u) {
    unsigned b = (u & 0x80000000u) ? (u & 0x7FFFFFFFu) : ~u;
    return __uint_as_float(b);
}

// ---------- init: zero scalars + segment buffers (rerun every call) ----------
__global__ void k_init(float* out01, unsigned* m_u, float* denom) {
    int i = blockIdx.x * blockDim.x + threadIdx.x;
    if (i < N_CELLS) { m_u[i] = 0u; denom[i] = 0.f; }
    if (i == 0) { out01[0] = 0.f; out01[1] = 0.f; }
}

// ---------- pre-swizzle B operands into WMMA fragment layout (f16) ----------
// frag index: tid = ((kt*CT + ct)*32 + lane)*16 + h
// lane<16:  col n = ct*16+lane,    K = kt*32 + (h<8 ? h : h+8)
// lane>=16: col n = ct*16+lane-16, K = kt*32 + 8 + (h<8 ? h : h+8)
__global__ void k_prep_frags(const float* __restrict__ Mu, const float* __restrict__ Var,
                             const float* __restrict__ lw, const float* __restrict__ rw,
                             _Float16* B1f, _Float16* B2f, _Float16* Glf, _Float16* Grf) {
    int tid = blockIdx.x * blockDim.x + threadIdx.x;
    if (tid >= KT_TILES * CT_TILES * 32 * 16) return;
    int h  = tid & 15;
    int L  = (tid >> 4) & 31;
    int ct = (tid >> 9) & 3;
    int kt = tid >> 11;
    int k  = kt * 32 + (L < 16 ? 0 : 8) + (h < 8 ? h : h + 8);
    int n  = ct * 16 + (L & 15);
    float b1 = 0.f, b2 = 0.f;
    if (k < N_GENES && n < N_CLASSES) {
        float iv = 1.0f / Var[n * N_GENES + k];
        b1 = iv;
        b2 = Mu[n * N_GENES + k] * iv;
    }
    B1f[tid] = (_Float16)b1;
    B2f[tid] = (_Float16)b2;
    if (ct == 0) {  // HID=8 fits in first column tile
        int gi = kt * 512 + L * 16 + h;
        float gvl = 0.f, gvr = 0.f;
        if (k < N_GENES && n < HID) {
            gvl = lw[n * N_GENES + k];
            gvr = rw[n * N_GENES + k];
        }
        Glf[gi] = (_Float16)gvl;
        Grf[gi] = (_Float16)gvr;
    }
}

// ---------- t3[c] = sum_g Mu^2 / Var (padded to 64 with zeros) ----------
__global__ void k_t3(const float* __restrict__ Mu, const float* __restrict__ Var, float* t3) {
    int c = threadIdx.x;
    if (c >= 64) return;
    float s = 0.f;
    if (c < N_CLASSES) {
        for (int g = 0; g < N_GENES; ++g) {
            float m = Mu[c * N_GENES + g];
            s += m * m / Var[c * N_GENES + g];
        }
    }
    t3[c] = s;
}

// ---------- row softmax of W -> P (one wave32 per row, 2 elems/lane) ----------
__global__ void k_softmax(const float* __restrict__ W, float* __restrict__ P) {
    int row  = blockIdx.x * 8 + (threadIdx.x >> 5);
    int lane = threadIdx.x & 31;
    const float* w = W + (long)row * N_CLASSES;
    float v0 = w[lane];                                     // lane 0..31 < 50
    float v1 = (lane + 32 < N_CLASSES) ? w[lane + 32] : -3.0e38f;
    float mx = fmaxf(v0, v1);
    for (int off = 16; off > 0; off >>= 1) mx = fmaxf(mx, __shfl_xor(mx, off, 32));
    float e0 = __expf(v0 - mx);
    float e1 = (lane + 32 < N_CLASSES) ? __expf(v1 - mx) : 0.f;
    float s = e0 + e1;
    for (int off = 16; off > 0; off >>= 1) s += __shfl_xor(s, off, 32);
    float inv = 1.0f / s;
    P[(long)row * N_CLASSES + lane] = e0 * inv;
    if (lane + 32 < N_CLASSES) P[(long)row * N_CLASSES + lane + 32] = e1 * inv;
}

// ---------- fused WMMA GEMM: t1,t2 (4 col tiles each) + gl,gr; reads X once ----------
__global__ void __launch_bounds__(128)
k_fused_gemm(const float* __restrict__ X, const float* __restrict__ S,
             const float* __restrict__ P,
             const _Float16* __restrict__ B1f, const _Float16* __restrict__ B2f,
             const _Float16* __restrict__ Glf, const _Float16* __restrict__ Grf,
             const float* __restrict__ t3,
             const float* __restrict__ lb, const float* __restrict__ rb,
             float* __restrict__ gl, float* __restrict__ gr, float* out_ll) {
    int wave = threadIdx.x >> 5;
    int lane = threadIdx.x & 31;
    int tile = blockIdx.x * 4 + wave;
    if (tile >= ROW_TILES) return;
    int rowbase = tile * 16;
    long myrow  = rowbase + (lane & 15);
    int  klo    = (lane < 16) ? 0 : 8;
    const float* xrow = X + myrow * N_GENES;

    v8f acc1[CT_TILES], acc2[CT_TILES], accl, accr;
    #pragma unroll
    for (int ct = 0; ct < CT_TILES; ++ct)
        #pragma unroll
        for (int j = 0; j < 8; ++j) { acc1[ct][j] = 0.f; acc2[ct][j] = 0.f; }
    #pragma unroll
    for (int j = 0; j < 8; ++j) { accl[j] = 0.f; accr[j] = 0.f; }

    for (int kt = 0; kt < KT_TILES; ++kt) {
        // A fragments: row myrow, K runs [kt*32+klo, +8) and [kt*32+klo+16, +8)
        v16h a, asq;
        float fs[16];
        if (kt < KT_TILES - 1) {
            const float4* p0 = (const float4*)(xrow + kt * 32 + klo);
            const float4* p1 = (const float4*)(xrow + kt * 32 + klo + 16);
            float4 f0 = p0[0], f1 = p0[1], f2 = p1[0], f3 = p1[1];
            fs[0]=f0.x; fs[1]=f0.y; fs[2]=f0.z; fs[3]=f0.w;
            fs[4]=f1.x; fs[5]=f1.y; fs[6]=f1.z; fs[7]=f1.w;
            fs[8]=f2.x; fs[9]=f2.y; fs[10]=f2.z; fs[11]=f2.w;
            fs[12]=f3.x; fs[13]=f3.y; fs[14]=f3.z; fs[15]=f3.w;
            // prefetch next K tile of this row (streams ~1KB ahead)
            __builtin_prefetch(xrow + (kt + 1) * 32 + klo, 0, 3);
        } else {
            // tail (K 480..511, valid < 500) — fully uniform, no exec divergence:
            // run1 (k = 480+klo .. 487+klo) is always in-bounds for both halves.
            const float4* p0 = (const float4*)(xrow + 480 + klo);
            float4 f0 = p0[0], f1 = p0[1];
            // run2: only lanes with klo==0 have valid data (k=496..499); load the
            // clamped in-bounds float4 at k=496 for everyone, select by lane half.
            const float4* p2 = (const float4*)(xrow + 496);
            float4 f2 = p2[0];
            bool lo = (lane < 16);
            fs[0]=f0.x; fs[1]=f0.y; fs[2]=f0.z; fs[3]=f0.w;
            fs[4]=f1.x; fs[5]=f1.y; fs[6]=f1.z; fs[7]=f1.w;
            fs[8]  = lo ? f2.x : 0.f;
            fs[9]  = lo ? f2.y : 0.f;
            fs[10] = lo ? f2.z : 0.f;
            fs[11] = lo ? f2.w : 0.f;
            fs[12] = 0.f; fs[13] = 0.f; fs[14] = 0.f; fs[15] = 0.f;
        }
        #pragma unroll
        for (int h = 0; h < 16; ++h) {
            float v = fs[h];
            a[h]   = (_Float16)v;
            asq[h] = (_Float16)(v * v);
        }
        #pragma unroll
        for (int ct = 0; ct < CT_TILES; ++ct) {
            v16h b1 = *(const v16h*)(B1f + (((long)kt * CT_TILES + ct) * 32 + lane) * 16);
            v16h b2 = *(const v16h*)(B2f + (((long)kt * CT_TILES + ct) * 32 + lane) * 16);
            acc1[ct] = __builtin_amdgcn_wmma_f32_16x16x32_f16(false, asq, false, b1,
                                                              (short)0, acc1[ct], false, false);
            acc2[ct] = __builtin_amdgcn_wmma_f32_16x16x32_f16(false, a,   false, b2,
                                                              (short)0, acc2[ct], false, false);
        }
        v16h bl = *(const v16h*)(Glf + ((long)kt * 32 + lane) * 16);
        v16h br = *(const v16h*)(Grf + ((long)kt * 32 + lane) * 16);
        accl = __builtin_amdgcn_wmma_f32_16x16x32_f16(false, a, false, bl, (short)0, accl, false, false);
        accr = __builtin_amdgcn_wmma_f32_16x16x32_f16(false, a, false, br, (short)0, accr, false, false);
    }

    // D layout: lane<16 -> (M=v, N=lane); lane>=16 -> (M=v+8, N=lane-16)
    float partial = 0.f;
    int ncol = lane & 15;
    #pragma unroll
    for (int v = 0; v < 8; ++v) {
        long r = rowbase + ((lane < 16) ? v : v + 8);
        float s = S[r];
        #pragma unroll
        for (int ct = 0; ct < CT_TILES; ++ct) {
            int n = ct * 16 + ncol;
            float F = -0.5f * (acc1[ct][v] - 2.0f * s * acc2[ct][v] + s * s * t3[n]);
            if (n < N_CLASSES) partial += P[r * N_CLASSES + n] * F;
        }
        if (ncol < HID) {
            gl[r * HID + ncol] = accl[v] + lb[ncol];
            gr[r * HID + ncol] = accr[v] + rb[ncol];
        }
    }
    for (int off = 16; off > 0; off >>= 1) partial += __shfl_xor(partial, off, 32);
    if (lane == 0) atomicAdd(out_ll, partial * (1.0f / (float)N_CELLS));
}

// ---------- edge pass 1: score + segment max over dst ----------
__global__ void k_edge_score(const int* __restrict__ ei,
                             const float* __restrict__ gl, const float* __restrict__ gr,
                             const float* __restrict__ att,
                             float* __restrict__ score, unsigned* __restrict__ m_u) {
    int e = blockIdx.x * blockDim.x + threadIdx.x;
    if (e >= N_EDGES) return;
    int s = ei[e], d = ei[N_EDGES + e];
    const float4* ps = (const float4*)(gl + (long)s * HID);
    const float4* pd = (const float4*)(gr + (long)d * HID);
    float4 a0 = ps[0], a1 = ps[1], b0 = pd[0], b1 = pd[1];
    float v[8] = { a0.x + b0.x, a0.y + b0.y, a0.z + b0.z, a0.w + b0.w,
                   a1.x + b1.x, a1.y + b1.y, a1.z + b1.z, a1.w + b1.w };
    float sc = 0.f;
    #pragma unroll
    for (int j = 0; j < HID; ++j) {
        float t = v[j];
        t = (t > 0.f) ? t : NEG_SLOPE * t;
        sc += att[j] * t;
    }
    score[e] = sc;
    atomicMax(&m_u[d], enc_f(sc));
}

// ---------- edge pass 2: denom[d] += exp(score - m[d]) ----------
__global__ void k_edge_denom(const int* __restrict__ ei, const float* __restrict__ score,
                             const unsigned* __restrict__ m_u, float* __restrict__ denom) {
    int e = blockIdx.x * blockDim.x + threadIdx.x;
    if (e >= N_EDGES) return;
    int d = ei[N_EDGES + e];
    float ex = __expf(score[e] - dec_f(m_u[d]));
    atomicAdd(&denom[d], ex);
}

// ---------- edge pass 3: ce -= alpha * (P[src] . log(P[dst]+1e-8)) / N ----------
__global__ void k_edge_ce(const int* __restrict__ ei, const float* __restrict__ score,
                          const unsigned* __restrict__ m_u, const float* __restrict__ denom,
                          const float* __restrict__ P, float* out_ce) {
    int e = blockIdx.x * blockDim.x + threadIdx.x;
    float partial = 0.f;
    if (e < N_EDGES) {
        int s = ei[e], d = ei[N_EDGES + e];
        float alpha = __expf(score[e] - dec_f(m_u[d])) / denom[d];
        const float2* ps = (const float2*)(P + (long)s * N_CLASSES);  // 8B aligned rows
        const float2* pd = (const float2*)(P + (long)d * N_CLASSES);
        float dot = 0.f;
        #pragma unroll
        for (int c = 0; c < N_CLASSES / 2; ++c) {
            float2 pa = ps[c], pb = pd[c];
            dot += pa.x * __logf(pb.x + 1e-8f) + pa.y * __logf(pb.y + 1e-8f);
        }
        partial = alpha * dot;
    }
    for (int off = 16; off > 0; off >>= 1) partial += __shfl_xor(partial, off, 32);
    if ((threadIdx.x & 31) == 0) atomicAdd(out_ce, -partial * (1.0f / (float)N_CELLS));
}

extern "C" void kernel_launch(void* const* d_in, const int* in_sizes, int n_in,
                              void* d_out, int out_size, void* d_ws, size_t ws_size,
                              hipStream_t stream) {
    const float* X   = (const float*)d_in[0];
    const float* Mu  = (const float*)d_in[1];
    const float* Var = (const float*)d_in[2];
    const float* W   = (const float*)d_in[3];
    const float* S   = (const float*)d_in[4];
    const int*   ei  = (const int*)  d_in[5];
    const float* lw  = (const float*)d_in[6];
    const float* lbb = (const float*)d_in[7];
    const float* rw  = (const float*)d_in[8];
    const float* rbb = (const float*)d_in[9];
    const float* att = (const float*)d_in[10];

    float* out  = (float*)d_out;      // [ll, ce, P...]
    float* outP = out + 2;

    // workspace carve (256B aligned slabs)
    char* w = (char*)d_ws;
    auto carve = [&](size_t bytes) -> char* {
        char* p = w;
        w += (bytes + 255) & ~(size_t)255;
        return p;
    };
    _Float16* B1f  = (_Float16*)carve((size_t)KT_TILES * CT_TILES * 32 * 16 * sizeof(_Float16));
    _Float16* B2f  = (_Float16*)carve((size_t)KT_TILES * CT_TILES * 32 * 16 * sizeof(_Float16));
    _Float16* Glf  = (_Float16*)carve((size_t)KT_TILES * 32 * 16 * sizeof(_Float16));
    _Float16* Grf  = (_Float16*)carve((size_t)KT_TILES * 32 * 16 * sizeof(_Float16));
    float*    t3   = (float*)   carve(64 * sizeof(float));
    float*    gl   = (float*)   carve((size_t)N_CELLS * HID * sizeof(float));
    float*    gr   = (float*)   carve((size_t)N_CELLS * HID * sizeof(float));
    float*    scr  = (float*)   carve((size_t)N_EDGES * sizeof(float));
    unsigned* m_u  = (unsigned*)carve((size_t)N_CELLS * sizeof(unsigned));
    float*    den  = (float*)   carve((size_t)N_CELLS * sizeof(float));

    k_init<<<(N_CELLS + 255) / 256, 256, 0, stream>>>(out, m_u, den);
    k_prep_frags<<<(KT_TILES * CT_TILES * 32 * 16 + 255) / 256, 256, 0, stream>>>(
        Mu, Var, lw, rw, B1f, B2f, Glf, Grf);
    k_t3<<<1, 64, 0, stream>>>(Mu, Var, t3);
    k_softmax<<<N_CELLS / 8, 256, 0, stream>>>(W, outP);
    k_fused_gemm<<<(ROW_TILES + 3) / 4, 128, 0, stream>>>(
        X, S, outP, B1f, B2f, Glf, Grf, t3, lbb, rbb, gl, gr, out);
    k_edge_score<<<(N_EDGES + 255) / 256, 256, 0, stream>>>(ei, gl, gr, att, scr, m_u);
    k_edge_denom<<<(N_EDGES + 255) / 256, 256, 0, stream>>>(ei, scr, m_u, den);
    k_edge_ce<<<(N_EDGES + 255) / 256, 256, 0, stream>>>(ei, scr, m_u, den, outP, out + 1);
}
